// GraphAttentionLayer_1047972020767
// MI455X (gfx1250) — matrix operand order repaired and checked by reading, not compiled, not run on hardware
//
#include <hip/hip_runtime.h>
#include <hip/hip_bf16.h>

typedef __attribute__((ext_vector_type(16))) _Float16 v16h;
typedef __attribute__((ext_vector_type(8)))  float    v8f;

#define NROW   8192
#define INF_   512
#define OUTF   256
#define GALPHA 0.2f

static __device__ __forceinline__ unsigned ballot32(bool p) {
#if __has_builtin(__builtin_amdgcn_ballot_w32)
  return (unsigned)__builtin_amdgcn_ballot_w32(p);
#else
  return (unsigned)__ballot(p);
#endif
}

// ---------------------------------------------------------------------------
// Kernel 1: Wh = h @ W^T + b   (f32 in, f16 WMMA, f32 accumulate)
// Block: 256 thr = 8 waves. Wave = 16 rows x 64 cols. Block = 32 rows x 256.
// ---------------------------------------------------------------------------
__global__ __launch_bounds__(256) void k_gemm_wh(const float* __restrict__ h,
                                                 const float* __restrict__ W,
                                                 const float* __restrict__ Wb,
                                                 float* __restrict__ Wh) {
  const int tid  = threadIdx.x;
  const int lane = tid & 31, wave = tid >> 5;
  const int l16  = lane & 15, hs = lane >> 4;
  const int row0 = blockIdx.x * 32 + (wave >> 2) * 16;
  const int n0   = (wave & 3) * 64;
  const int arow = row0 + l16;
  const int basek = hs * 8;   // A layout: lanes 0-15 K{0..7,16..23}; 16-31 K{8..15,24..31}

  v8f acc[4] = {};

  for (int k0 = 0; k0 < INF_; k0 += 32) {
    v16h a;
    const float* hp = h + (size_t)arow * INF_ + k0 + basek;
#pragma unroll
    for (int t = 0; t < 8; ++t) a[t] = (_Float16)hp[t];
#pragma unroll
    for (int t = 0; t < 8; ++t) a[8 + t] = (_Float16)hp[16 + t];
#pragma unroll
    for (int nt = 0; nt < 4; ++nt) {
      const int ncol = n0 + nt * 16 + l16;
      // B layout: lanes 0-15 hold N=l16, K {0..7,16..23}; lanes 16-31 hold K {8..15,24..31}.
      const float* wp = W + (size_t)ncol * INF_ + k0 + hs * 8;
      v16h b;
#pragma unroll
      for (int t = 0; t < 16; ++t) b[t] = (_Float16)wp[t < 8 ? t : t + 8];
      acc[nt] = __builtin_amdgcn_wmma_f32_16x16x32_f16(false, a, false, b,
                                                       (short)0, acc[nt], false, false);
    }
  }
#pragma unroll
  for (int nt = 0; nt < 4; ++nt) {
    const int col  = n0 + nt * 16 + l16;
    const float bias = Wb[col];
#pragma unroll
    for (int v = 0; v < 8; ++v) {  // C layout: VGPR v -> M = v + 8*hs, N = l16
      const int row = row0 + v + hs * 8;
      Wh[(size_t)row * OUTF + col] = acc[nt][v] + bias;
    }
  }
}

// ---------------------------------------------------------------------------
// Kernel 2a: f1[i] = Wh[i]·a1, f2[i] = Wh[i]·a2  (one wave per row)
// ---------------------------------------------------------------------------
__global__ __launch_bounds__(256) void k_fvec(const float* __restrict__ Wh,
                                              const float* __restrict__ aw,
                                              float* __restrict__ f1,
                                              float* __restrict__ f2) {
  const int lane = threadIdx.x & 31;
  const int row  = blockIdx.x * 8 + (threadIdx.x >> 5);
  const float* wr = Wh + (size_t)row * OUTF;
  float s1 = 0.f, s2 = 0.f;
#pragma unroll
  for (int t = 0; t < 8; ++t) {
    const int c = lane + 32 * t;
    const float w = wr[c];
    s1 += w * aw[c];
    s2 += w * aw[OUTF + c];
  }
#pragma unroll
  for (int m = 16; m >= 1; m >>= 1) {
    s1 += __shfl_xor(s1, m, 32);
    s2 += __shfl_xor(s2, m, 32);
  }
  if (lane == 0) { f1[row] = s1; f2[row] = s2; }
}

// ---------------------------------------------------------------------------
// Kernel 2b: WhT[o][i] = (f16)Wh[i][o]  (tiled LDS transpose, 32x32)
// ---------------------------------------------------------------------------
__global__ __launch_bounds__(256) void k_transpose(const float* __restrict__ Wh,
                                                   _Float16* __restrict__ WhT) {
  __shared__ float tile[32][33];
  const int tx = threadIdx.x, ty = threadIdx.y;         // blockDim (32, 8)
  const int i0 = blockIdx.y * 32, o0 = blockIdx.x * 32;
#pragma unroll
  for (int m = 0; m < 4; ++m)
    tile[ty + 8 * m][tx] = Wh[(size_t)(i0 + ty + 8 * m) * OUTF + o0 + tx];
  __syncthreads();
#pragma unroll
  for (int m = 0; m < 4; ++m)
    WhT[(size_t)(o0 + ty + 8 * m) * NROW + i0 + tx] = (_Float16)tile[tx][ty + 8 * m];
}

// ---------------------------------------------------------------------------
// Kernel 3: single pass over adj (256 MB): pack bitmask (8 MB, L2-resident)
// and rowmax[i] = max f2[j] over neighbors. One block per row.
// ---------------------------------------------------------------------------
__global__ __launch_bounds__(256) void k_mask(const int* __restrict__ adj,
                                              const float* __restrict__ f2,
                                              unsigned* __restrict__ maskbits,
                                              float* __restrict__ rowmax) {
  const int i   = blockIdx.x;
  const int tid = threadIdx.x;
  const int lane = tid & 31, wave = tid >> 5;
  const int* arow = adj + (size_t)i * NROW;
  float mx = -3.0e38f;
#pragma unroll 4
  for (int c = 0; c < NROW / 256; ++c) {
    const int j = c * 256 + tid;
    const bool nb = arow[j] > 0;
    const unsigned word = ballot32(nb);
    if (lane == 0) maskbits[(size_t)i * (NROW / 32) + c * 8 + wave] = word;
    if (nb) { const float v = f2[j]; mx = v > mx ? v : mx; }
  }
  __shared__ float red[256];
  red[tid] = mx;
  __syncthreads();
  for (int s = 128; s >= 1; s >>= 1) {
    if (tid < s) { const float o = red[tid + s]; if (o > red[tid]) red[tid] = o; }
    __syncthreads();
  }
  if (tid == 0) rowmax[i] = red[0];
}

// ---------------------------------------------------------------------------
// Kernel 4: fused masked-softmax @ Wh with ELU epilogue (flash-style).
// Block: 256 thr = 8 waves = 4 row-groups x 2 col-strips -> 64 rows x 256 cols.
// Wave: 16 rows x 128 cols -> 8 WMMA accumulators; P tile built in registers
// per the 16-bit A-matrix lane layout; softmax denominator accumulated inline.
// ---------------------------------------------------------------------------
__global__ __launch_bounds__(256) void k_attn(const _Float16* __restrict__ WhT,
                                              const float* __restrict__ f1,
                                              const float* __restrict__ f2,
                                              const float* __restrict__ rowmax,
                                              const unsigned* __restrict__ maskbits,
                                              const float* __restrict__ ab_ptr,
                                              float* __restrict__ out) {
  __shared__ float f2s[NROW];         // 32 KB of the 320 KB WGP LDS
  __shared__ float sArr[8][16];
  const int tid = threadIdx.x;
  for (int t = tid; t < NROW; t += 256) f2s[t] = f2[t];
  __syncthreads();

  const int lane = tid & 31, wave = tid >> 5;
  const int l16  = lane & 15, hs = lane >> 4;
  const int row0 = blockIdx.x * 64 + (wave >> 1) * 16;
  const int n0   = (wave & 1) * 128;
  const int i    = row0 + l16;
  const float ab  = ab_ptr[0];
  const float f1v = f1[i];
  const float tm  = f1v + ab + rowmax[i];
  const float m   = tm > 0.f ? tm : GALPHA * tm;   // row softmax max (lrelu monotone)
  const unsigned* mrow = maskbits + (size_t)i * (NROW / 32);
  const int basek = hs * 16;

  v8f acc[8] = {};
  float ssum = 0.f;

  for (int j0 = 0; j0 < NROW; j0 += 32) {
    const unsigned word = mrow[j0 >> 5];
    v16h a;
#pragma unroll
    for (int t = 0; t < 16; ++t) {
      const int k = basek + t;   // {0..15} or {16..31}
      const float e0 = f1v + f2s[j0 + k] + ab;
      const float lr = e0 > 0.f ? e0 : GALPHA * e0;
      const float p  = ((word >> k) & 1u) ? __expf(lr - m) : 0.f;
      ssum += p;
      a[t] = (_Float16)p;
    }
#pragma unroll
    for (int nt = 0; nt < 8; ++nt) {
      const int ncol = n0 + nt * 16 + l16;
      const v16h b = *(const v16h*)(WhT + (size_t)ncol * NROW + j0 + hs * 16);
      acc[nt] = __builtin_amdgcn_wmma_f32_16x16x32_f16(false, a, false, b,
                                                       (short)0, acc[nt], false, false);
    }
  }

  // Combine the two K-half partials of each row, broadcast s per row via LDS.
  ssum += __shfl_xor(ssum, 16, 32);
  if (lane < 16) sArr[wave][l16] = ssum;
  __syncthreads();

#pragma unroll
  for (int v = 0; v < 8; ++v) {
    const int row = row0 + v + hs * 8;
    float s = sArr[wave][v + hs * 8];
    s = s > 0.f ? s : 1.f;               // guard: isolated row (ref -> uniform; see note)
    const float inv = 1.f / s;
#pragma unroll
    for (int nt = 0; nt < 8; ++nt) {
      const int col = n0 + nt * 16 + l16;
      const float x = acc[nt][v] * inv;
      out[(size_t)row * OUTF + col] = x > 0.f ? x : __expf(x) - 1.f;
    }
  }
}

// ---------------------------------------------------------------------------
extern "C" void kernel_launch(void* const* d_in, const int* in_sizes, int n_in,
                              void* d_out, int out_size, void* d_ws, size_t ws_size,
                              hipStream_t stream) {
  const float* h   = (const float*)d_in[0];
  const int*   adj = (const int*)  d_in[1];
  const float* Ww  = (const float*)d_in[2];
  const float* Wb  = (const float*)d_in[3];
  const float* aw  = (const float*)d_in[4];
  const float* ab  = (const float*)d_in[5];
  float* out = (float*)d_out;

  // Workspace layout (~20.1 MB total):
  char* ws = (char*)d_ws;
  float*    Wh   = (float*)   (ws);              //  8 MB: Wh f32 [8192][256]
  _Float16* WhT  = (_Float16*)(ws +  8388608);   //  4 MB: Wh^T f16 [256][8192]
  float*    f1   = (float*)   (ws + 12582912);   // 32 KB
  float*    f2   = (float*)   (ws + 12615680);   // 32 KB
  float*    rmax = (float*)   (ws + 12648448);   // 32 KB
  unsigned* mask = (unsigned*)(ws + 12681216);   //  8 MB: adjacency bitmask

  k_gemm_wh <<<NROW / 32, 256, 0, stream>>>(h, Ww, Wb, Wh);
  k_fvec    <<<NROW / 8,  256, 0, stream>>>(Wh, aw, f1, f2);
  k_transpose<<<dim3(OUTF / 32, NROW / 32), dim3(32, 8), 0, stream>>>(Wh, WhT);
  k_mask    <<<NROW, 256, 0, stream>>>(adj, f2, mask, rmax);
  k_attn    <<<NROW / 64, 256, 0, stream>>>(WhT, f1, f2, rmax, mask, ab, out);
}
